// NaylisAttention_42726334660835
// MI455X (gfx1250) — matrix operand order repaired and checked by reading, not compiled, not run on hardware
//
#include <hip/hip_runtime.h>
#include <cstddef>
#include <cstdint>

// ---------------------------------------------------------------------------
// Types for CDNA5 WMMA (wave32): 16x16x32 bf16, fp32 accumulate
// ---------------------------------------------------------------------------
typedef __attribute__((ext_vector_type(16))) __bf16 v16bf;
typedef __attribute__((ext_vector_type(8)))  float  v8f;
typedef __attribute__((ext_vector_type(4)))  uint32_t u32x4;
typedef __attribute__((ext_vector_type(8)))  int32_t  i32x8;
typedef __attribute__((ext_vector_type(4)))  int32_t  i32x4;

__device__ inline v8f zero8() {
  v8f z = {0.f, 0.f, 0.f, 0.f, 0.f, 0.f, 0.f, 0.f};
  return z;
}

__device__ inline v8f wmma_bf16(v16bf a, v16bf b, v8f c) {
  // (neg_a, A, neg_b, B, c_mod, C, reuse_a, reuse_b)
  return __builtin_amdgcn_wmma_f32_16x16x32_bf16(false, a, false, b, (short)0, c,
                                                 false, false);
}

// A-matrix 16x32 bf16 fragment (row-major source, leading dim ld).
// Layout: lanes 0-15 -> M=lane, K 0..15; lanes 16-31 -> K 16..31 (matches B loaders)
__device__ inline v16bf load_a16x32(const __bf16* __restrict__ p, int ld, int lane) {
  const int m  = lane & 15;
  const int kb = (lane >> 4) << 4;   // 0 or 16
  const __bf16* r = p + (size_t)m * ld;
  v16bf a;
#pragma unroll
  for (int pi = 0; pi < 8; ++pi) {
    const int k = kb + (pi << 1);  // kb+2*pi
    a[2 * pi]     = r[k];
    a[2 * pi + 1] = r[k + 1];
  }
  return a;
}

// B-matrix 32x16: element (k,n) = p[n*ld + k]  (i.e. "N rows of K", GEMM with W^T)
__device__ inline v16bf load_b32x16_nmajor(const __bf16* __restrict__ p, int ld, int lane) {
  const int n  = lane & 15;
  const int kb = (lane >> 4) << 4;   // 0 or 16
  const __bf16* r = p + (size_t)n * ld + kb;
  v16bf b;
#pragma unroll
  for (int e = 0; e < 16; ++e) b[e] = r[e];
  return b;
}

// B-matrix 32x16: element (k,n) = p[k*ld + n]  (K-major rows, used for V)
__device__ inline v16bf load_b32x16_kmajor(const __bf16* __restrict__ p, int ld, int lane) {
  const int n  = lane & 15;
  const int kb = (lane >> 4) << 4;
  v16bf b;
#pragma unroll
  for (int e = 0; e < 16; ++e) b[e] = p[(size_t)(kb + e) * ld + n];
  return b;
}

// 16-lane half-wave reductions (C-fragment rows live across 16-lane halves)
__device__ inline float halfmax(float x) {
  x = fmaxf(x, __shfl_xor(x, 1));
  x = fmaxf(x, __shfl_xor(x, 2));
  x = fmaxf(x, __shfl_xor(x, 4));
  x = fmaxf(x, __shfl_xor(x, 8));
  return x;
}
__device__ inline float halfsum(float x) {
  x += __shfl_xor(x, 1);
  x += __shfl_xor(x, 2);
  x += __shfl_xor(x, 4);
  x += __shfl_xor(x, 8);
  return x;
}

// ---------------------------------------------------------------------------
// Tensor Data Mover: async 2-D tile (rows x cols bf16) Global -> LDS.
// D# packing per CDNA5 ISA ch.8.3/8.4 (group0: count/lds/global/type,
// group1: data_size=2B, tensor dims, tile dims, row stride).
// Toolchain uses the 6-arg builtin: (g0, g1, g2, g3, g4, cpol).
// ---------------------------------------------------------------------------
__device__ inline void tdm_load_2d(__bf16* lds_dst, const __bf16* gsrc,
                                   int ld_elems, int rows, int cols) {
  const uint64_t ga = (uint64_t)(uintptr_t)gsrc;
  const uint32_t lds_off = (uint32_t)(uintptr_t)lds_dst;  // flat[31:0] == LDS byte offset
  u32x4 g0;
  g0.x = 1u;                                         // count=1 (valid descriptor)
  g0.y = lds_off;                                    // lds_addr
  g0.z = (uint32_t)ga;                               // global_addr[31:0]
  g0.w = (uint32_t)((ga >> 32) & 0x01FFFFFFu) | (2u << 30);  // addr[56:32] | type=2
  const uint32_t td0 = (uint32_t)ld_elems;           // tensor_dim0 (elements)
  const uint32_t td1 = (uint32_t)rows;               // tensor_dim1
  const uint64_t st0 = (uint64_t)ld_elems;           // tensor_dim0_stride (elements)
  i32x8 g1;
  g1[0] = (int)(1u << 16);                           // workgroup_mask=0, data_size=1 (2B)
  g1[1] = (int)((td0 & 0xFFFFu) << 16);              // tensor_dim0[15:0]
  g1[2] = (int)((td0 >> 16) | ((td1 & 0xFFFFu) << 16));
  g1[3] = (int)((td1 >> 16) | ((uint32_t)cols << 16));   // tile_dim0 = cols
  g1[4] = (int)(uint32_t)rows;                       // tile_dim1 = rows, tile_dim2=0
  g1[5] = (int)(uint32_t)st0;                        // stride0[31:0]
  g1[6] = (int)((uint32_t)(st0 >> 32) & 0xFFFFu);    // stride0[47:32], stride1=0
  g1[7] = 0;
  i32x4 g2 = {0, 0, 0, 0};
  i32x4 g3 = {0, 0, 0, 0};
  i32x8 g4 = {0, 0, 0, 0, 0, 0, 0, 0};
  __builtin_amdgcn_tensor_load_to_lds(g0, g1, g2, g3, g4, 0);
}

// ---------------------------------------------------------------------------
// Problem constants
// ---------------------------------------------------------------------------
#define S_LEN 2048
#define EDIM  2048
#define HQ    16
#define KVH   4
#define DHEAD 128
#define KVROW 512     // KVH*DHEAD
#define HG    12
#define RRANK 32
#define RROW  384     // HG*RRANK
#define NSYM  4

// ---------------------------------------------------------------------------
// fp32 -> bf16 conversion
// ---------------------------------------------------------------------------
__global__ void cvt_f32_bf16(const float* __restrict__ in, __bf16* __restrict__ out,
                             int n) {
  int i = blockIdx.x * blockDim.x + threadIdx.x;
  if (i < n) out[i] = (__bf16)in[i];
}

// ---------------------------------------------------------------------------
// GEMM: C(MxN,f32) = A(MxK,bf16) @ B(NxK,bf16)^T.
// Workgroup = 8 waves -> 128x128 tile; K staged in 32-deep LDS chunks,
// double-buffered via TDM async loads (wave 0 drives the DMA engine,
// s_wait_tensorcnt + barrier publish each buffer). Per wave: 32x64 output.
// LDS: 2 * (128x32 A + 128x32 B) bf16 = 32 KB.
// ---------------------------------------------------------------------------
#define BM 128
#define BN 128
#define BK 32

__global__ __launch_bounds__(256)
void gemm_bf16_tdm(const __bf16* __restrict__ A, const __bf16* __restrict__ B,
                   float* __restrict__ C, int M, int N, int K) {
  __shared__ __bf16 Asm[2][BM][BK];
  __shared__ __bf16 Bsm[2][BN][BK];
  const int lane  = threadIdx.x & 31;
  const int wave  = threadIdx.x >> 5;
  const int m0 = blockIdx.y * BM;
  const int n0 = blockIdx.x * BN;
  const int mwave = (wave & 3) * 32;   // 4 waves along M
  const int nwave = (wave >> 2) * 64;  // 2 waves along N
  const int nk = K / BK;

  if (wave == 0) {  // prologue: chunk 0 in flight
    tdm_load_2d(&Asm[0][0][0], A + (size_t)m0 * K, K, BM, BK);
    tdm_load_2d(&Bsm[0][0][0], B + (size_t)n0 * K, K, BN, BK);
  }

  v8f acc[2][4];
#pragma unroll
  for (int mt = 0; mt < 2; ++mt)
#pragma unroll
    for (int nt = 0; nt < 4; ++nt) acc[mt][nt] = zero8();

  for (int c = 0; c < nk; ++c) {
    const int cur = c & 1;
    if (wave == 0) {
      if (c + 1 < nk) {  // post next chunk, then wait for current pair only
        tdm_load_2d(&Asm[cur ^ 1][0][0], A + (size_t)m0 * K + (size_t)(c + 1) * BK,
                    K, BM, BK);
        tdm_load_2d(&Bsm[cur ^ 1][0][0], B + (size_t)n0 * K + (size_t)(c + 1) * BK,
                    K, BN, BK);
        __builtin_amdgcn_s_wait_tensorcnt(2);
      } else {
        __builtin_amdgcn_s_wait_tensorcnt(0);
      }
    }
    __syncthreads();  // publish LDS buffer `cur` to all waves

    v16bf a0 = load_a16x32(&Asm[cur][mwave][0],      BK, lane);
    v16bf a1 = load_a16x32(&Asm[cur][mwave + 16][0], BK, lane);
#pragma unroll
    for (int nt = 0; nt < 4; ++nt) {
      v16bf b = load_b32x16_nmajor(&Bsm[cur][nwave + nt * 16][0], BK, lane);
      acc[0][nt] = wmma_bf16(a0, b, acc[0][nt]);
      acc[1][nt] = wmma_bf16(a1, b, acc[1][nt]);
    }
    __syncthreads();  // all reads done before next TDM overwrite
  }

  // C fragment: VGPR v -> row +v +8*(lane>=16), col n = lane&15
  const int mb = m0 + mwave + ((lane >> 4) << 3);
  const int nc = lane & 15;
#pragma unroll
  for (int mt = 0; mt < 2; ++mt)
#pragma unroll
    for (int nt = 0; nt < 4; ++nt)
#pragma unroll
      for (int v = 0; v < 8; ++v)
        C[(size_t)(mb + mt * 16 + v) * N + n0 + nwave + nt * 16 + nc] =
            acc[mt][nt][v];
}

// ---------------------------------------------------------------------------
// Per-head RMSNorm + RoPE, fp32 in -> bf16 out.  grid=(S, heads), block=128
// ---------------------------------------------------------------------------
__global__ void norm_rope_kernel(const float* __restrict__ in,
                                 const float* __restrict__ w,
                                 __bf16* __restrict__ out, int rowlen) {
  const int s = blockIdx.x, h = blockIdx.y, d = threadIdx.x;
  const float* row = in + (size_t)s * rowlen + h * DHEAD;
  float val = row[d];
  __shared__ float red[DHEAD];
  red[d] = val * val;
  __syncthreads();
  for (int off = 64; off > 0; off >>= 1) {
    if (d < off) red[d] += red[d + off];
    __syncthreads();
  }
  const float rsq = rsqrtf(red[0] * (1.0f / DHEAD) + 1e-6f);
  const float vn = val * rsq * w[d];
  const int j = d & 63;
  const float inv = __powf(10000.0f, -(float)(2 * j) * (1.0f / DHEAD));
  float sn, cs;
  __sincosf((float)s * inv, &sn, &cs);
  const int partner = (d < 64) ? d + 64 : d - 64;
  const float pvn = row[partner] * rsq * w[partner];
  const float rot = (d < 64) ? -pvn : pvn;
  out[(size_t)s * rowlen + h * DHEAD + d] = (__bf16)(vn * cs + rot * sn);
}

// ---------------------------------------------------------------------------
// Fused flash attention with on-the-fly rank-32 graph bias.
// grid=(S/16, H), block=32 (one wave per 16-query strip).
// ---------------------------------------------------------------------------
__global__ __launch_bounds__(32)
void flash_bias_kernel(const __bf16* __restrict__ Q, const __bf16* __restrict__ K,
                       const __bf16* __restrict__ V, const __bf16* __restrict__ Rq,
                       const __bf16* __restrict__ Rk, const float* __restrict__ gscale,
                       float* __restrict__ O) {
  const int lane  = threadIdx.x;
  const int qt    = blockIdx.x;
  const int h     = blockIdx.y;
  const int kvh   = h >> 2;                 // rep = H/KVH = 4
  const int qbase = qt * 16;
  const float scale = 0.08838834764831845f; // 1/sqrt(128)
  const bool has_bias = (h < HG);
  const bool sym      = (h < NSYM);
  const float gs = has_bias ? gscale[h] : 0.0f;

  // Q fragments (16 rows x 128 = 4 A-frags), loaded once
  v16bf qa[4];
#pragma unroll
  for (int db = 0; db < 4; ++db)
    qa[db] = load_a16x32(Q + (size_t)qbase * EDIM + h * DHEAD + db * 32, EDIM, lane);

  v16bf ra = {}, rka = {};
  if (has_bias) ra  = load_a16x32(Rq + (size_t)qbase * RROW + h * RRANK, RROW, lane);
  if (sym)      rka = load_a16x32(Rk + (size_t)qbase * RROW + h * RRANK, RROW, lane);

  float mrow[8], lrow[8];
  v8f o[8];
#pragma unroll
  for (int v = 0; v < 8; ++v) { mrow[v] = -3.0e38f; lrow[v] = 0.0f; o[v] = zero8(); }

  __shared__ __bf16 pls[16][32];
  const int rowoff = (lane >> 4) << 3;  // 0 or 8
  const int ncol   = lane & 15;

  const int nchunk = (qbase + 16 + 31) >> 5;  // 32-key chunks up to the diagonal
  for (int c = 0; c < nchunk; ++c) {
    const int kb = c << 5;
    // ---- S = Q K^T (two 16x16 tiles over 32 keys) -------------------------
    v8f s0 = zero8(), s1 = zero8();
#pragma unroll
    for (int db = 0; db < 4; ++db) {
      v16bf b0 = load_b32x16_nmajor(K + (size_t)kb * KVROW + kvh * DHEAD + db * 32, KVROW, lane);
      v16bf b1 = load_b32x16_nmajor(K + (size_t)(kb + 16) * KVROW + kvh * DHEAD + db * 32, KVROW, lane);
      s0 = wmma_bf16(qa[db], b0, s0);
      s1 = wmma_bf16(qa[db], b1, s1);
    }
    // ---- rank-32 graph bias: one WMMA per 16-key tile ---------------------
    if (has_bias) {
      v16bf rb0 = load_b32x16_nmajor(Rk + (size_t)kb * RROW + h * RRANK, RROW, lane);
      v16bf rb1 = load_b32x16_nmajor(Rk + (size_t)(kb + 16) * RROW + h * RRANK, RROW, lane);
      v8f b0 = wmma_bf16(ra, rb0, zero8());
      v8f b1 = wmma_bf16(ra, rb1, zero8());
      if (sym) {  // (bias + bias^T)/2 computed locally: Rk[s].Rq[t] term
        v16bf qb0 = load_b32x16_nmajor(Rq + (size_t)kb * RROW + h * RRANK, RROW, lane);
        v16bf qb1 = load_b32x16_nmajor(Rq + (size_t)(kb + 16) * RROW + h * RRANK, RROW, lane);
        v8f t0 = wmma_bf16(rka, qb0, zero8());
        v8f t1 = wmma_bf16(rka, qb1, zero8());
#pragma unroll
        for (int v = 0; v < 8; ++v) {
          b0[v] = 0.5f * (b0[v] + t0[v]);
          b1[v] = 0.5f * (b1[v] + t1[v]);
        }
      }
#pragma unroll
      for (int v = 0; v < 8; ++v) {
        s0[v] = s0[v] * scale + b0[v] * gs;
        s1[v] = s1[v] * scale + b1[v] * gs;
      }
    } else {
#pragma unroll
      for (int v = 0; v < 8; ++v) { s0[v] *= scale; s1[v] *= scale; }
    }
    // ---- causal mask + online softmax (rows live across 16-lane halves) ---
    float p0[8], p1[8];
#pragma unroll
    for (int v = 0; v < 8; ++v) {
      const int row = qbase + rowoff + v;
      if (kb + ncol > row)      s0[v] = -3.0e38f;
      if (kb + 16 + ncol > row) s1[v] = -3.0e38f;
      float rm = halfmax(fmaxf(s0[v], s1[v]));
      const float mnew  = fmaxf(mrow[v], rm);
      const float alpha = __expf(mrow[v] - mnew);
      mrow[v] = mnew;
      p0[v] = __expf(s0[v] - mnew);
      p1[v] = __expf(s1[v] - mnew);
      const float rs = halfsum(p0[v] + p1[v]);
      lrow[v] = lrow[v] * alpha + rs;
#pragma unroll
      for (int t = 0; t < 8; ++t) o[t][v] *= alpha;
    }
    // ---- re-layout P (C-frag) -> A-frag through 1KB LDS -------------------
#pragma unroll
    for (int v = 0; v < 8; ++v) {
      pls[rowoff + v][ncol]      = (__bf16)p0[v];
      pls[rowoff + v][ncol + 16] = (__bf16)p1[v];
    }
    __syncthreads();
    v16bf pa = load_a16x32(&pls[0][0], 32, lane);
    __syncthreads();
    // ---- O += P @ V (16x32 @ 32x128) --------------------------------------
#pragma unroll
    for (int t = 0; t < 8; ++t) {
      v16bf vb = load_b32x16_kmajor(V + (size_t)kb * KVROW + kvh * DHEAD + t * 16, KVROW, lane);
      o[t] = wmma_bf16(pa, vb, o[t]);
    }
  }
  // ---- normalize and store ------------------------------------------------
#pragma unroll
  for (int t = 0; t < 8; ++t) {
#pragma unroll
    for (int v = 0; v < 8; ++v) {
      const int row = qbase + rowoff + v;
      O[(size_t)row * EDIM + h * DHEAD + t * 16 + ncol] = o[t][v] / lrow[v];
    }
  }
}

// ---------------------------------------------------------------------------
// Host-side launch
// ---------------------------------------------------------------------------
extern "C" void kernel_launch(void* const* d_in, const int* in_sizes, int n_in,
                              void* d_out, int out_size, void* d_ws, size_t ws_size,
                              hipStream_t stream) {
  (void)in_sizes; (void)n_in; (void)out_size; (void)ws_size;
  const float* x        = (const float*)d_in[0];
  const float* Wq       = (const float*)d_in[1];
  const float* Wk       = (const float*)d_in[2];
  const float* Wv       = (const float*)d_in[3];
  const float* Wo       = (const float*)d_in[4];
  const float* q_norm_w = (const float*)d_in[5];
  const float* k_norm_w = (const float*)d_in[6];
  const float* Wrq      = (const float*)d_in[7];
  const float* Wrk      = (const float*)d_in[8];
  const float* gscale   = (const float*)d_in[9];
  float* out = (float*)d_out;

  // workspace bump allocator (256B aligned), total 100 MiB
  char* wp = (char*)d_ws;
  auto alloc = [&](size_t bytes) -> void* {
    void* r = (void*)wp;
    wp += (bytes + 255) & ~(size_t)255;
    return r;
  };
  const size_t nx   = (size_t)S_LEN * EDIM;     // 4M
  const size_t nkv  = (size_t)S_LEN * KVROW;    // 1M
  const size_t nr   = (size_t)S_LEN * RROW;     // 0.75M
  const size_t nwq  = (size_t)EDIM * EDIM;
  const size_t nwkv = (size_t)KVROW * EDIM;
  const size_t nwr  = (size_t)RROW * EDIM;

  __bf16* x_bf   = (__bf16*)alloc(nx * 2);
  __bf16* Wq_bf  = (__bf16*)alloc(nwq * 2);
  __bf16* Wk_bf  = (__bf16*)alloc(nwkv * 2);
  __bf16* Wv_bf  = (__bf16*)alloc(nwkv * 2);
  __bf16* Wo_bf  = (__bf16*)alloc(nwq * 2);
  __bf16* Wrq_bf = (__bf16*)alloc(nwr * 2);
  __bf16* Wrk_bf = (__bf16*)alloc(nwr * 2);
  float*  Qf     = (float*)alloc(nx * 4);
  float*  Kf     = (float*)alloc(nkv * 4);
  float*  Vf     = (float*)alloc(nkv * 4);
  float*  Rqf    = (float*)alloc(nr * 4);
  float*  Rkf    = (float*)alloc(nr * 4);
  __bf16* Q_bf   = (__bf16*)alloc(nx * 2);
  __bf16* K_bf   = (__bf16*)alloc(nkv * 2);
  __bf16* V_bf   = (__bf16*)alloc(nkv * 2);
  __bf16* Rq_bf  = (__bf16*)alloc(nr * 2);
  __bf16* Rk_bf  = (__bf16*)alloc(nr * 2);
  float*  attnf  = (float*)alloc(nx * 4);
  __bf16* attnbf = (__bf16*)alloc(nx * 2);

  auto cvt = [&](const float* in, __bf16* o, size_t n) {
    cvt_f32_bf16<<<(int)((n + 255) / 256), 256, 0, stream>>>(in, o, (int)n);
  };
  auto gemm = [&](const __bf16* A, const __bf16* B, float* C, int M, int N, int K) {
    gemm_bf16_tdm<<<dim3(N / BN, M / BM), 256, 0, stream>>>(A, B, C, M, N, K);
  };

  // 1) bf16 conversions of activations + weights
  cvt(x, x_bf, nx);
  cvt(Wq, Wq_bf, nwq);
  cvt(Wk, Wk_bf, nwkv);
  cvt(Wv, Wv_bf, nwkv);
  cvt(Wo, Wo_bf, nwq);
  cvt(Wrq, Wrq_bf, nwr);
  cvt(Wrk, Wrk_bf, nwr);

  // 2) projections (all x @ W^T); N = 2048/512/512/384/384, all /128
  gemm(x_bf, Wq_bf, Qf, S_LEN, EDIM, EDIM);
  gemm(x_bf, Wk_bf, Kf, S_LEN, KVROW, EDIM);
  gemm(x_bf, Wv_bf, Vf, S_LEN, KVROW, EDIM);
  gemm(x_bf, Wrq_bf, Rqf, S_LEN, RROW, EDIM);
  gemm(x_bf, Wrk_bf, Rkf, S_LEN, RROW, EDIM);

  // 3) RMSNorm + RoPE for Q/K (fp32 math), V/Rq/Rk straight conversion
  norm_rope_kernel<<<dim3(S_LEN, HQ), DHEAD, 0, stream>>>(Qf, q_norm_w, Q_bf, EDIM);
  norm_rope_kernel<<<dim3(S_LEN, KVH), DHEAD, 0, stream>>>(Kf, k_norm_w, K_bf, KVROW);
  cvt(Vf, V_bf, nkv);
  cvt(Rqf, Rq_bf, nr);
  cvt(Rkf, Rk_bf, nr);

  // 4) fused causal flash attention + rank-32 graph bias
  flash_bias_kernel<<<dim3(S_LEN / 16, HQ), 32, 0, stream>>>(Q_bf, K_bf, V_bf,
                                                             Rq_bf, Rk_bf, gscale,
                                                             attnf);

  // 5) output projection
  cvt(attnf, attnbf, nx);
  gemm(attnbf, Wo_bf, out, S_LEN, EDIM, EDIM);
}
